// MiniMaxText01AttentionType0_64244120813650
// MI455X (gfx1250) — compile-verified, matrix operand order checked by reading, not run on hardware
//
#include <hip/hip_runtime.h>

#define BB    2
#define NSEQ  4096
#define HID   2048
#define HH    16
#define DD    128
#define BLK   256
#define NBK   16
#define M_TOT (BB*NSEQ)   /* 8192 */
#define NQKV  (3*HH*DD)   /* 6144 */

typedef __attribute__((ext_vector_type(16))) __bf16 v16bf;
typedef __attribute__((ext_vector_type(8)))  float  v8f;
typedef unsigned short __attribute__((address_space(3))) lds_ushort;

union FragU {
  v16bf v;
  uint4 q[2];
  unsigned short u[16];
};

// Native bf16 converts (gfx1250 has bf16 VALU/convert hardware).
__device__ __forceinline__ unsigned short f2bf(float f) {
  union { __bf16 b; unsigned short u; } x;
  x.b = (__bf16)f;
  return x.u;
}
__device__ __forceinline__ float bf2f(unsigned short h) {
  union { __bf16 b; unsigned short u; } x;
  x.u = h;
  return (float)x.b;
}
__device__ __forceinline__ v8f wmma_bf16(v16bf a, v16bf b, v8f c) {
  return __builtin_amdgcn_wmma_f32_16x16x32_bf16(false, a, false, b, (short)0, c, false, false);
}
__device__ __forceinline__ v8f v8f_zero() {
  v8f z = {0.f,0.f,0.f,0.f,0.f,0.f,0.f,0.f};
  return z;
}

// ---------------------------------------------------------------------------
// Generic 128x128-tile bf16 WMMA GEMM: C = A(MxK) * W(NxK)^T with fused
// epilogues. 256 threads = 8 waves in a 4(M) x 2(N) grid; each wave owns a
// 32x64 patch = 2x4 tiles of 16x16 f32 accumulators. K staged 32 at a time
// through LDS with f32->bf16 conversion on the store.
//   MODE 0: SiLU + scatter into q/k/v (b,h,blk,row,d) bf16 buffers
//   MODE 1: gate = sigmoid(C); act = bf16(gate * attn*rms*norm_w)
//   MODE 2: A is bf16 (act); async Global->LDS stage; plain f32 store to out
// ---------------------------------------------------------------------------
template<int MODE>
__global__ __launch_bounds__(256)
void gemm_bf16_kernel(const float* __restrict__ A32,
                      const unsigned short* __restrict__ A16,
                      const float* __restrict__ W,
                      unsigned short* __restrict__ qws,
                      unsigned short* __restrict__ kws,
                      unsigned short* __restrict__ vws,
                      const float* __restrict__ attn,
                      const float* __restrict__ rscale,
                      const float* __restrict__ normw,
                      unsigned short* __restrict__ act,
                      float* __restrict__ outp,
                      int K)
{
  __shared__ unsigned short Al[128 * 32];   // [m][kk]
  __shared__ unsigned short Bl[128 * 32];   // [n][kk]

  const int tid  = threadIdx.x;
  const int lane = tid & 31, w = tid >> 5;
  const int llo  = lane & 15, lhi = lane >> 4;
  const int wm   = w & 3,  wn = w >> 2;
  const int mBase = blockIdx.y * 128;
  const int nBase = blockIdx.x * 128;

  v8f acc[2][4];
#pragma unroll
  for (int rt = 0; rt < 2; rt++)
#pragma unroll
    for (int ct = 0; ct < 4; ct++) acc[rt][ct] = v8f_zero();

  const int ldRow = tid >> 1;
  const int ldCol = (tid & 1) * 16;

  for (int k0 = 0; k0 < K; k0 += 32) {
    // ---- stage A tile (128x32) ----
    if (MODE == 2) {
      // A already bf16: pure byte move -> CDNA5 async Global->LDS DMA path.
      const unsigned short* src = A16 + (size_t)(mBase + ldRow) * K + k0 + ldCol;
      lds_ushort* l0 = (lds_ushort*)&Al[ldRow * 32 + ldCol];
      lds_ushort* l1 = (lds_ushort*)&Al[ldRow * 32 + ldCol + 8];
      asm volatile(
        "global_load_async_to_lds_b128 %0, %2, off\n\t"
        "global_load_async_to_lds_b128 %1, %3, off"
        :
        : "v"(l0), "v"(l1), "v"(src), "v"(src + 8)
        : "memory");
      __builtin_prefetch(src + 32, 0, 1);
    } else {
      const float* src = A32 + (size_t)(mBase + ldRow) * K + k0 + ldCol;
      FragU t;
#pragma unroll
      for (int e = 0; e < 16; e++) t.u[e] = f2bf(src[e]);
      *(uint4*)(&Al[ldRow * 32 + ldCol])     = t.q[0];
      *(uint4*)(&Al[ldRow * 32 + ldCol + 8]) = t.q[1];
      __builtin_prefetch(src + 32, 0, 1);
    }
    // ---- stage B tile (rows n, 32 kk) ----
    {
      const float* src = W + (size_t)(nBase + ldRow) * K + k0 + ldCol;
      FragU t;
#pragma unroll
      for (int e = 0; e < 16; e++) t.u[e] = f2bf(src[e]);
      *(uint4*)(&Bl[ldRow * 32 + ldCol])     = t.q[0];
      *(uint4*)(&Bl[ldRow * 32 + ldCol + 8]) = t.q[1];
      __builtin_prefetch(src + 32, 0, 1);
    }
    if (MODE == 2)
      asm volatile("s_wait_asynccnt 0" ::: "memory");  // async LDS writes visible
    __syncthreads();

    FragU af[2], bf[4];
#pragma unroll
    for (int rt = 0; rt < 2; rt++) {
      const unsigned short* p = &Al[(32 * wm + 16 * rt + llo) * 32];
      af[rt].q[0] = *(const uint4*)(p + 8 * lhi);
      af[rt].q[1] = *(const uint4*)(p + 16 + 8 * lhi);
    }
#pragma unroll
    for (int ct = 0; ct < 4; ct++) {
      const unsigned short* p = &Bl[(64 * wn + 16 * ct + llo) * 32];
      bf[ct].q[0] = *(const uint4*)(p + 8 * lhi);
      bf[ct].q[1] = *(const uint4*)(p + 16 + 8 * lhi);
    }
#pragma unroll
    for (int rt = 0; rt < 2; rt++)
#pragma unroll
      for (int ct = 0; ct < 4; ct++)
        acc[rt][ct] = wmma_bf16(af[rt].v, bf[ct].v, acc[rt][ct]);
    __syncthreads();
  }

  // ---- epilogue ----
#pragma unroll
  for (int rt = 0; rt < 2; rt++) {
#pragma unroll
    for (int ct = 0; ct < 4; ct++) {
#pragma unroll
      for (int r = 0; r < 8; r++) {
        float cv = acc[rt][ct][r];
        int m_g = mBase + 32 * wm + 16 * rt + r + 8 * lhi;
        int n_g = nBase + 64 * wn + 16 * ct + llo;
        if (MODE == 0) {
          float sv  = cv / (1.f + __expf(-cv));   // SiLU
          int hI   = n_g / 384;
          int rem  = n_g - hI * 384;
          int part = rem >> 7;
          int dc   = rem & 127;
          int bi   = m_g >> 12;
          int nr   = m_g & 4095;
          int ib   = nr >> 8;
          int rr   = nr & 255;
          unsigned short* dst = (part == 0) ? qws : (part == 1) ? kws : vws;
          dst[((((size_t)(bi * HH + hI)) * NBK + ib) * BLK + rr) * DD + dc] = f2bf(sv);
        } else if (MODE == 1) {
          float g = 1.f / (1.f + __expf(-cv));    // sigmoid gate
          float o = attn[(size_t)m_g * HID + n_g] * rscale[m_g] * normw[n_g];
          act[(size_t)m_g * HID + n_g] = f2bf(g * o);
        } else {
          outp[(size_t)m_g * HID + n_g] = cv;
        }
      }
    }
  }
}

// ---------------------------------------------------------------------------
// Lightning-attention block scan. One workgroup per (b,h); 8 waves each own a
// 32-row strip of the 256-row block and a 16-row strip of the kv^T state
// (f32 WMMA accumulators, persistent across the 16-block scan).
// ---------------------------------------------------------------------------
__global__ __launch_bounds__(256)
void lightning_attn_kernel(const unsigned short* __restrict__ qws,
                           const unsigned short* __restrict__ kws,
                           const unsigned short* __restrict__ vws,
                           const float* __restrict__ slope_rate,
                           float* __restrict__ attn)
{
  __shared__ unsigned short kvT[128 * 128];     // [dj][di] bf16 copy of state (32KB)
  __shared__ unsigned short kT[128 * 32];       // [di][key] chunk of k^T (8KB)
  __shared__ unsigned short vT[128 * 32];       // [dj][key] chunk of v^T (8KB)
  __shared__ unsigned short sx[8][32 * 32];     // per-wave C->A layout bounce (16KB)

  const int tid  = threadIdx.x;
  const int lane = tid & 31, w = tid >> 5;
  const int llo  = lane & 15, lhi = lane >> 4;
  const int bi   = blockIdx.x >> 4;
  const int h    = blockIdx.x & 15;
  const float slope  = slope_rate[h];
  const float bdecay = __expf(-slope * (float)BLK);

  v8f kvacc[8];                                  // kv^T rows [16w,16w+16) x 128
#pragma unroll
  for (int ct = 0; ct < 8; ct++) kvacc[ct] = v8f_zero();

  const size_t headBase = ((size_t)(bi * HH + h)) * NBK * BLK * DD;

  for (int ib = 0; ib < NBK; ib++) {
    const unsigned short* qb = qws + headBase + (size_t)ib * BLK * DD;
    const unsigned short* kb = kws + headBase + (size_t)ib * BLK * DD;
    const unsigned short* vb = vws + headBase + (size_t)ib * BLK * DD;

    // ---- phase 1: publish current kv^T state to LDS as bf16 ----
#pragma unroll
    for (int ct = 0; ct < 8; ct++)
#pragma unroll
      for (int r = 0; r < 8; r++) {
        int dj = 16 * w + r + 8 * lhi;
        int di = 16 * ct + llo;
        kvT[dj * 128 + di] = f2bf(kvacc[ct][r]);
      }
    __syncthreads();
    // decay old state before accumulating this block's contribution
#pragma unroll
    for (int ct = 0; ct < 8; ct++)
#pragma unroll
      for (int r = 0; r < 8; r++) kvacc[ct][r] *= bdecay;

    // ---- phase 2: load q fragments; inter-block part out = q @ kv ----
    FragU qf[2][4];
#pragma unroll
    for (int rt = 0; rt < 2; rt++) {
      const unsigned short* p = qb + (32 * w + 16 * rt + llo) * DD;
#pragma unroll
      for (int kt = 0; kt < 4; kt++) {
        qf[rt][kt].q[0] = *(const uint4*)(p + 32 * kt + 8 * lhi);
        qf[rt][kt].q[1] = *(const uint4*)(p + 32 * kt + 16 + 8 * lhi);
      }
    }
    v8f outa[2][8];
#pragma unroll
    for (int rt = 0; rt < 2; rt++)
#pragma unroll
      for (int ct = 0; ct < 8; ct++) outa[rt][ct] = v8f_zero();

#pragma unroll
    for (int ct = 0; ct < 8; ct++) {
      const unsigned short* p = &kvT[(16 * ct + llo) * 128];
#pragma unroll
      for (int kt = 0; kt < 4; kt++) {
        FragU bk;
        bk.q[0] = *(const uint4*)(p + 32 * kt + 8 * lhi);
        bk.q[1] = *(const uint4*)(p + 32 * kt + 16 + 8 * lhi);
#pragma unroll
        for (int rt = 0; rt < 2; rt++)
          outa[rt][ct] = wmma_bf16(qf[rt][kt].v, bk.v, outa[rt][ct]);
      }
    }
    // inter part is row-scaled by q_decay = exp(-s*(m+1)) (decay is per row,
    // so apply it to the product instead of the q operand)
#pragma unroll
    for (int rt = 0; rt < 2; rt++) {
#pragma unroll
      for (int r = 0; r < 8; r++) {
        float rdec = __expf(-slope * (float)(32 * w + 16 * rt + r + 8 * lhi + 1));
#pragma unroll
        for (int ct = 0; ct < 8; ct++) outa[rt][ct][r] *= rdec;
      }
    }

    // ---- phase 3: intra part + state update, 8 chunks of 32 keys ----
    for (int c = 0; c < 8; c++) {
      { // cooperative transpose-stage of k,v chunk into LDS
        int key = tid >> 3;
        int d0  = (tid & 7) * 16;
        FragU tk, tv;
        tk.q[0] = *(const uint4*)(kb + (32 * c + key) * DD + d0);
        tk.q[1] = *(const uint4*)(kb + (32 * c + key) * DD + d0 + 8);
        tv.q[0] = *(const uint4*)(vb + (32 * c + key) * DD + d0);
        tv.q[1] = *(const uint4*)(vb + (32 * c + key) * DD + d0 + 8);
#pragma unroll
        for (int e = 0; e < 16; e++) {
          kT[(d0 + e) * 32 + key] = tk.u[e];
          vT[(d0 + e) * 32 + key] = tv.u[e];
        }
      }
      __syncthreads();

      if (c <= w) {     // causal: wave rows [32w,32w+32) vs keys [32c,32c+32)
        // s = q @ k^T for this chunk (32x32, f32)
        v8f s[2][2];
#pragma unroll
        for (int rt = 0; rt < 2; rt++)
#pragma unroll
          for (int cn = 0; cn < 2; cn++) s[rt][cn] = v8f_zero();
#pragma unroll
        for (int kt = 0; kt < 4; kt++) {
#pragma unroll
          for (int cn = 0; cn < 2; cn++) {
            const unsigned short* p = kb + (32 * c + 16 * cn + llo) * DD;
            FragU bk;
            bk.q[0] = *(const uint4*)(p + 32 * kt + 8 * lhi);
            bk.q[1] = *(const uint4*)(p + 32 * kt + 16 + 8 * lhi);
#pragma unroll
            for (int rt = 0; rt < 2; rt++)
              s[rt][cn] = wmma_bf16(qf[rt][kt].v, bk.v, s[rt][cn]);
          }
        }
        // decay mask, bounce C-layout -> A-layout through per-wave LDS
#pragma unroll
        for (int rt = 0; rt < 2; rt++)
#pragma unroll
          for (int cn = 0; cn < 2; cn++)
#pragma unroll
            for (int r = 0; r < 8; r++) {
              int ml = 32 * w + 16 * rt + r + 8 * lhi;
              int nl = 32 * c + 16 * cn + llo;
              float v = s[rt][cn][r];
              v = (ml >= nl) ? v * __expf(-slope * (float)(ml - nl)) : 0.f;
              sx[w][(16 * rt + r + 8 * lhi) * 32 + 16 * cn + llo] = f2bf(v);
            }
        asm volatile("s_wait_dscnt 0" ::: "memory");   // wave-local RAW on LDS
        FragU sa[2];
#pragma unroll
        for (int rt = 0; rt < 2; rt++) {
          const unsigned short* p = &sx[w][(16 * rt + llo) * 32];
          sa[rt].q[0] = *(const uint4*)(p + 8 * lhi);
          sa[rt].q[1] = *(const uint4*)(p + 16 + 8 * lhi);
        }
        // out += s @ v_chunk
#pragma unroll
        for (int ct = 0; ct < 8; ct++) {
          const unsigned short* p = &vT[(16 * ct + llo) * 32];
          FragU bv;
          bv.q[0] = *(const uint4*)(p + 8 * lhi);
          bv.q[1] = *(const uint4*)(p + 16 + 8 * lhi);
#pragma unroll
          for (int rt = 0; rt < 2; rt++)
            outa[rt][ct] = wmma_bf16(sa[rt].v, bv.v, outa[rt][ct]);
        }
      }

      // kv^T strip update: kvT[16w..16w+16, :] += (v^T row-strip * k_decay) @ k
      {
        const unsigned short* p = &vT[(16 * w + llo) * 32];
        FragU av;
        av.q[0] = *(const uint4*)(p + 8 * lhi);
        av.q[1] = *(const uint4*)(p + 16 + 8 * lhi);
#pragma unroll
        for (int e = 0; e < 16; e++) {
          int nk = e + 8 * lhi + ((e >= 8) ? 8 : 0);   // key index within chunk
          float kd = __expf(-slope * (float)(255 - (32 * c + nk)));
          av.u[e] = f2bf(bf2f(av.u[e]) * kd);
        }
#pragma unroll
        for (int ct = 0; ct < 8; ct++) {
          const unsigned short* p2 = &kT[(16 * ct + llo) * 32];
          FragU bkk;
          bkk.q[0] = *(const uint4*)(p2 + 8 * lhi);
          bkk.q[1] = *(const uint4*)(p2 + 16 + 8 * lhi);
          kvacc[ct] = wmma_bf16(av.v, bkk.v, kvacc[ct]);
        }
      }
      __syncthreads();
    }

    // ---- write this block's output (b, n, h*d layout, f32) ----
#pragma unroll
    for (int rt = 0; rt < 2; rt++)
#pragma unroll
      for (int ct = 0; ct < 8; ct++)
#pragma unroll
        for (int r = 0; r < 8; r++) {
          int gr  = ib * BLK + 32 * w + 16 * rt + r + 8 * lhi;
          int col = h * DD + 16 * ct + llo;
          attn[((size_t)bi * NSEQ + gr) * HID + col] = outa[rt][ct][r];
        }
  }
}

// ---------------------------------------------------------------------------
// RMSNorm row scale: rscale[m] = rsqrt(mean(out^2) + eps)
// ---------------------------------------------------------------------------
__global__ __launch_bounds__(256)
void rms_kernel(const float* __restrict__ attn, float* __restrict__ rscale)
{
  __shared__ float red[256];
  int row = blockIdx.x;
  const float* p = attn + (size_t)row * HID;
  float s = 0.f;
  for (int i = threadIdx.x; i < HID; i += 256) { float v = p[i]; s += v * v; }
  red[threadIdx.x] = s;
  __syncthreads();
  for (int off = 128; off > 0; off >>= 1) {
    if (threadIdx.x < off) red[threadIdx.x] += red[threadIdx.x + off];
    __syncthreads();
  }
  if (threadIdx.x == 0)
    rscale[row] = rsqrtf(red[0] / (float)HID + 1.1920929e-7f);
}

// ---------------------------------------------------------------------------
extern "C" void kernel_launch(void* const* d_in, const int* in_sizes, int n_in,
                              void* d_out, int out_size, void* d_ws, size_t ws_size,
                              hipStream_t stream)
{
  const float* x     = (const float*)d_in[0];
  const float* slope = (const float*)d_in[1];
  const float* Wqkv  = (const float*)d_in[2];
  const float* Wg    = (const float*)d_in[3];
  const float* normw = (const float*)d_in[4];
  const float* Wo    = (const float*)d_in[5];
  float* out = (float*)d_out;

  // workspace layout (bytes):
  //   qws  bf16  [0,            33554432)
  //   kws  bf16  [33554432,     67108864)
  //   vws  bf16  [67108864,    100663296)
  //   attn f32   [100663296,   167772160)
  //   rscale f32 [167772160,   167804928)
  //   act  bf16  [167804928,   201359360)
  char* ws = (char*)d_ws;
  unsigned short* qws    = (unsigned short*)(ws);
  unsigned short* kws    = (unsigned short*)(ws + (size_t)33554432);
  unsigned short* vws    = (unsigned short*)(ws + (size_t)67108864);
  float*          attn   = (float*)(ws + (size_t)100663296);
  float*          rscale = (float*)(ws + (size_t)167772160);
  unsigned short* act    = (unsigned short*)(ws + (size_t)167804928);

  dim3 blk(256);
  gemm_bf16_kernel<0><<<dim3(NQKV / 128, M_TOT / 128), blk, 0, stream>>>(
      x, nullptr, Wqkv, qws, kws, vws, nullptr, nullptr, nullptr, nullptr, nullptr, HID);
  lightning_attn_kernel<<<dim3(BB * HH), blk, 0, stream>>>(qws, kws, vws, slope, attn);
  rms_kernel<<<dim3(M_TOT), blk, 0, stream>>>(attn, rscale);
  gemm_bf16_kernel<1><<<dim3(HID / 128, M_TOT / 128), blk, 0, stream>>>(
      x, nullptr, Wg, nullptr, nullptr, nullptr, attn, rscale, normw, act, nullptr, HID);
  gemm_bf16_kernel<2><<<dim3(HID / 128, M_TOT / 128), blk, 0, stream>>>(
      nullptr, act, Wo, nullptr, nullptr, nullptr, nullptr, nullptr, nullptr, nullptr, out, HID);
}
